// OptimizedLSTMModel_45595372814665
// MI455X (gfx1250) — compile-verified, matrix operand order checked by reading, not run on hardware
//
#include <hip/hip_runtime.h>
#include <hip/hip_bf16.h>

// ---------------------------------------------------------------------------
// Bi-LSTM (B=2048, S=128, I=128, H=64) + MLP head, for gfx1250 (CDNA5).
// bf16 WMMA (v_wmma_f32_16x16x32_bf16) with f32 accumulate.
// Kernel 1: one block per (16-row batch tile x direction) = 256 blocks,
//           16 waves x 16 gate columns each; weights resident in VGPRs;
//           double-buffered x staging software-pipelined around the WMMAs;
//           sched_barrier forces batched ds_loads (one exposed LDS latency).
// Kernel 2: MLP head with two WMMA GEMMs.
// ---------------------------------------------------------------------------

typedef __attribute__((ext_vector_type(16))) __bf16 v16bf;
typedef __attribute__((ext_vector_type(8)))  float  v8f;

#define B_TOT 2048
#define SEQ   128
#define IN    128
#define HID   64
#define GATES 256     // 4*HID
#define FC1   128
#define NCLS  16

#if __has_builtin(__builtin_amdgcn_sched_barrier)
#define SCHED_FENCE() __builtin_amdgcn_sched_barrier(0)
#else
#define SCHED_FENCE()
#endif

static __device__ __forceinline__ float fast_sigmoid(float v) {
  return 1.0f / (1.0f + __expf(-v));
}
static __device__ __forceinline__ float fast_tanh(float v) {
  float e = __expf(2.0f * v);
  return (e - 1.0f) / (e + 1.0f);
}

// B-fragment (K=32 slice of a [N x K] row-major weight; rows = output cols).
// lane L: n = L&15, K-half selected by L/16; 16 contiguous K values.
static __device__ __forceinline__ v16bf load_b_frag(const float* __restrict__ W,
                                                    int ldw, int colbase, int kb,
                                                    int lane) {
  int n = lane & 15;
  const float* src = W + (size_t)(colbase + n) * ldw + kb + ((lane >> 4) << 4);
  v16bf r;
#pragma unroll
  for (int i = 0; i < 16; ++i) r[i] = (__bf16)src[i];
  return r;
}

// A-fragment (16 x 32 bf16 tile from LDS); stride (elements) keeps 32B align.
static __device__ __forceinline__ v16bf load_a_frag(const __bf16* base, int stride,
                                                    int kb, int lane) {
  int m = lane & 15;
  const __bf16* p = base + (size_t)m * stride + kb + ((lane >> 4) << 4);
  return *(const v16bf*)p;
}

static __device__ __forceinline__ v8f wmma_bf16(v16bf a, v16bf b, v8f c) {
  return __builtin_amdgcn_wmma_f32_16x16x32_bf16(false, a, false, b,
                                                 (short)0, c, false, false);
}

// ---------------------------------------------------------------------------
// Kernel 1: full LSTM pass for one 16-row batch tile, one direction.
// gridDim.x = (B/16)*2 ; blockIdx.x = tile*2 + dir ; blockDim = 512 (16 waves)
// ---------------------------------------------------------------------------
__global__ __launch_bounds__(512)
void lstm_dir_kernel(const float* __restrict__ x,
                     const float* __restrict__ Wih_f, const float* __restrict__ Whh_f,
                     const float* __restrict__ bih_f, const float* __restrict__ bhh_f,
                     const float* __restrict__ Wih_b, const float* __restrict__ Whh_b,
                     const float* __restrict__ bih_b, const float* __restrict__ bhh_b,
                     float* __restrict__ hcat)
{
  // Row strides are multiples of 32B (aligned ds_read_b128) and rotate banks.
  __shared__ __attribute__((aligned(32))) __bf16 xs[2][16][144]; // x_t (dbl buf)
  __shared__ __attribute__((aligned(32))) __bf16 hs[16][80];     // h state
  __shared__ float cs[16][64];                                   // cell state
  __shared__ float gs[16][264];                                  // gate preacts
  __shared__ float bs[GATES];                                    // bih + bhh

  const int tile = blockIdx.x >> 1;
  const int dir  = blockIdx.x & 1;
  const int row0 = tile * 16;
  const int w    = threadIdx.x >> 5;   // wave 0..15 -> gate cols [16w, 16w+16)
  const int lane = threadIdx.x & 31;
  const int colbase = w * 16;

  const float* Wih = dir ? Wih_b : Wih_f;
  const float* Whh = dir ? Whh_b : Whh_f;
  const float* bih = dir ? bih_b : bih_f;
  const float* bhh = dir ? bhh_b : bhh_f;

  // Resident weight fragments for this wave's 16 gate columns.
  v16bf bWih[4];
  v16bf bWhh[2];
#pragma unroll
  for (int ks = 0; ks < 4; ++ks)
    bWih[ks] = load_b_frag(Wih, IN, colbase, ks * 32, lane);
#pragma unroll
  for (int ks = 0; ks < 2; ++ks)
    bWhh[ks] = load_b_frag(Whh, HID, colbase, ks * 32, lane);

  // Bias to LDS (consumed in the element-wise stage).
  for (int idx = threadIdx.x; idx < GATES; idx += 512)
    bs[idx] = bih[idx] + bhh[idx];

  for (int idx = threadIdx.x; idx < 16 * HID; idx += 512) {
    int m = idx >> 6, j = idx & 63;
    hs[m][j] = (__bf16)0.0f;
    cs[m][j] = 0.0f;
  }

  // Prologue: stage x(t=0) into buffer 0.
  {
    int ts0 = dir ? (SEQ - 1) : 0;
    for (int idx = threadIdx.x; idx < 16 * IN; idx += 512) {
      int m = idx >> 7, k = idx & 127;
      xs[0][m][k] = (__bf16)x[((size_t)(row0 + m) * SEQ + ts0) * IN + k];
    }
  }
  __syncthreads();

  for (int t = 0; t < SEQ; ++t) {
    const int buf = t & 1;

    // Issue next timestep's global loads NOW (latency hidden behind WMMAs);
    // convert + store to the other LDS buffer after the matrix ops.
    float xstage[4];
    int   sm[4], sk[4];
    const bool have_next = (t + 1 < SEQ);
    if (have_next) {
      int tn = dir ? (SEQ - 2 - t) : (t + 1);
#pragma unroll
      for (int i = 0; i < 4; ++i) {
        int idx = threadIdx.x + i * 512;
        sm[i] = idx >> 7; sk[i] = idx & 127;
        xstage[i] = x[((size_t)(row0 + sm[i]) * SEQ + tn) * IN + sk[i]];
      }
    }
    if (t + 2 < SEQ && threadIdx.x < 128) {
      int tp = dir ? (SEQ - 3 - t) : (t + 2);
      int m = threadIdx.x >> 3, q = threadIdx.x & 7;
      __builtin_prefetch(&x[((size_t)(row0 + m) * SEQ + tp) * IN + q * 16], 0, 0);
    }

    // Preload ALL A-fragments (h first: it is the recurrence-critical input),
    // then fence the scheduler so the 12 ds_load_b128 issue back-to-back and
    // the WMMA chain pays a single exposed LDS latency.
    v16bf ah0 = load_a_frag(&hs[0][0],      80,  0 * 32, lane);
    v16bf ah1 = load_a_frag(&hs[0][0],      80,  1 * 32, lane);
    v16bf ax0 = load_a_frag(&xs[buf][0][0], 144, 0 * 32, lane);
    v16bf ax1 = load_a_frag(&xs[buf][0][0], 144, 1 * 32, lane);
    v16bf ax2 = load_a_frag(&xs[buf][0][0], 144, 2 * 32, lane);
    v16bf ax3 = load_a_frag(&xs[buf][0][0], 144, 3 * 32, lane);
    SCHED_FENCE();

    // gates = x_t @ Wih^T + h @ Whh^T   (zero-init accumulator, bias later)
    v8f acc = {};
    acc = wmma_bf16(ax0, bWih[0], acc);
    acc = wmma_bf16(ax1, bWih[1], acc);
    acc = wmma_bf16(ax2, bWih[2], acc);
    acc = wmma_bf16(ax3, bWih[3], acc);
    acc = wmma_bf16(ah0, bWhh[0], acc);
    acc = wmma_bf16(ah1, bWhh[1], acc);

    // Scatter gate pre-activations (C/D layout: vgpr r -> M, lane -> N).
    {
      int col = colbase + (lane & 15);
#pragma unroll
      for (int r = 0; r < 8; ++r) {
        int m = (lane < 16) ? r : (r + 8);
        gs[m][col] = acc[r];
      }
    }

    // Fill the other x buffer for t+1 (loads already in flight).
    if (have_next) {
#pragma unroll
      for (int i = 0; i < 4; ++i)
        xs[buf ^ 1][sm[i]][sk[i]] = (__bf16)xstage[i];
    }
    __syncthreads();

    // Fused cell update: i,f,g,o at cols j, 64+j, 128+j, 192+j.
    for (int idx = threadIdx.x; idx < 16 * HID; idx += 512) {
      int m = idx >> 6, j = idx & 63;
      float ig = fast_sigmoid(gs[m][j]       + bs[j]);
      float fg = fast_sigmoid(gs[m][64 + j]  + bs[64 + j]);
      float gg = fast_tanh   (gs[m][128 + j] + bs[128 + j]);
      float og = fast_sigmoid(gs[m][192 + j] + bs[192 + j]);
      float c  = fg * cs[m][j] + ig * gg;
      cs[m][j] = c;
      hs[m][j] = (__bf16)(og * fast_tanh(c));
    }
    __syncthreads();
  }

  // Final hidden -> hcat[B][128] (fwd: cols 0..63, bwd: cols 64..127)
  for (int idx = threadIdx.x; idx < 16 * HID; idx += 512) {
    int m = idx >> 6, j = idx & 63;
    hcat[(size_t)(row0 + m) * (2 * HID) + dir * HID + j] = (float)hs[m][j];
  }
}

// ---------------------------------------------------------------------------
// Kernel 2: out = relu(hcat @ W1^T + b1) @ W2^T + b2, one block per 16 rows.
// ---------------------------------------------------------------------------
__global__ __launch_bounds__(256)
void mlp_head_kernel(const float* __restrict__ hcat,
                     const float* __restrict__ W1, const float* __restrict__ b1,
                     const float* __restrict__ W2, const float* __restrict__ b2,
                     float* __restrict__ out)
{
  __shared__ __attribute__((aligned(32))) __bf16 as[16][144];
  __shared__ __attribute__((aligned(32))) __bf16 rs[16][144];
  const int row0 = blockIdx.x * 16;
  const int w    = threadIdx.x >> 5;
  const int lane = threadIdx.x & 31;

  for (int idx = threadIdx.x; idx < 16 * (2 * HID); idx += 256) {
    int m = idx >> 7, k = idx & 127;
    as[m][k] = (__bf16)hcat[(size_t)(row0 + m) * (2 * HID) + k];
  }
  __syncthreads();

  // FC1: wave w owns output cols [16w, 16w+16), K = 128.
  {
    v16bf bw[4];
#pragma unroll
    for (int ks = 0; ks < 4; ++ks)
      bw[ks] = load_b_frag(W1, 2 * HID, w * 16, ks * 32, lane);
    float bias = b1[w * 16 + (lane & 15)];
    v16bf a0 = load_a_frag(&as[0][0], 144, 0 * 32, lane);
    v16bf a1 = load_a_frag(&as[0][0], 144, 1 * 32, lane);
    v16bf a2 = load_a_frag(&as[0][0], 144, 2 * 32, lane);
    v16bf a3 = load_a_frag(&as[0][0], 144, 3 * 32, lane);
    SCHED_FENCE();
    v8f acc = {};
    acc = wmma_bf16(a0, bw[0], acc);
    acc = wmma_bf16(a1, bw[1], acc);
    acc = wmma_bf16(a2, bw[2], acc);
    acc = wmma_bf16(a3, bw[3], acc);
    int col = w * 16 + (lane & 15);
#pragma unroll
    for (int r = 0; r < 8; ++r) {
      int m = (lane < 16) ? r : (r + 8);
      rs[m][col] = (__bf16)fmaxf(acc[r] + bias, 0.0f);
    }
  }
  __syncthreads();

  // FC2: [16x128] x [128x16]; wave 0 only (uniform branch -> EXEC all ones).
  if (w == 0) {
    v16bf bw[4];
#pragma unroll
    for (int ks = 0; ks < 4; ++ks)
      bw[ks] = load_b_frag(W2, FC1, 0, ks * 32, lane);
    float bias = b2[lane & 15];
    v16bf a0 = load_a_frag(&rs[0][0], 144, 0 * 32, lane);
    v16bf a1 = load_a_frag(&rs[0][0], 144, 1 * 32, lane);
    v16bf a2 = load_a_frag(&rs[0][0], 144, 2 * 32, lane);
    v16bf a3 = load_a_frag(&rs[0][0], 144, 3 * 32, lane);
    SCHED_FENCE();
    v8f acc = {};
    acc = wmma_bf16(a0, bw[0], acc);
    acc = wmma_bf16(a1, bw[1], acc);
    acc = wmma_bf16(a2, bw[2], acc);
    acc = wmma_bf16(a3, bw[3], acc);
    int col = lane & 15;
#pragma unroll
    for (int r = 0; r < 8; ++r) {
      int m = (lane < 16) ? r : (r + 8);
      out[(size_t)(row0 + m) * NCLS + col] = acc[r] + bias;
    }
  }
}

// ---------------------------------------------------------------------------
extern "C" void kernel_launch(void* const* d_in, const int* in_sizes, int n_in,
                              void* d_out, int out_size, void* d_ws, size_t ws_size,
                              hipStream_t stream) {
  (void)in_sizes; (void)n_in; (void)out_size; (void)ws_size;
  const float* x     = (const float*)d_in[0];
  const float* Wih_f = (const float*)d_in[1];
  const float* Whh_f = (const float*)d_in[2];
  const float* bih_f = (const float*)d_in[3];
  const float* bhh_f = (const float*)d_in[4];
  const float* Wih_b = (const float*)d_in[5];
  const float* Whh_b = (const float*)d_in[6];
  const float* bih_b = (const float*)d_in[7];
  const float* bhh_b = (const float*)d_in[8];
  const float* W1    = (const float*)d_in[9];
  const float* b1    = (const float*)d_in[10];
  const float* W2    = (const float*)d_in[11];
  const float* b2    = (const float*)d_in[12];

  float* hcat = (float*)d_ws;            // B_TOT * 2H floats = 1 MiB scratch

  lstm_dir_kernel<<<dim3((B_TOT / 16) * 2), dim3(512), 0, stream>>>(
      x, Wih_f, Whh_f, bih_f, bhh_f, Wih_b, Whh_b, bih_b, bhh_b, hcat);

  mlp_head_kernel<<<dim3(B_TOT / 16), dim3(256), 0, stream>>>(
      hcat, W1, b1, W2, b2, (float*)d_out);
}
